// AnomalyDAE_Base_51685636440167
// MI455X (gfx1250) — matrix operand order repaired and checked by reading, not compiled, not run on hardware
//
#include <hip/hip_runtime.h>
#include <stdint.h>

#define N_NODES 16384
#define F_DIM   512
#define EMB     128
#define OUT_D   64
#define N_EDGE  524288
#define TOT_E   (N_EDGE + N_NODES)   // edges + self loops

typedef __attribute__((ext_vector_type(16))) __bf16 v16bf;
typedef __attribute__((ext_vector_type(8)))  float  v8f;

union FragBF { v16bf v; uint32_t u[8]; };

__device__ __forceinline__ uint16_t f2bf(float f) {
    uint32_t u = __float_as_uint(f);
    u += 0x7FFFu + ((u >> 16) & 1u);       // round-to-nearest-even
    return (uint16_t)(u >> 16);
}
__device__ __forceinline__ float bf2f(uint16_t h) {
    return __uint_as_float(((uint32_t)h) << 16);
}
// order-preserving float->uint key for atomic max
__device__ __forceinline__ unsigned int fkey(float f) {
    unsigned int u = __float_as_uint(f);
    return (u & 0x80000000u) ? ~u : (u | 0x80000000u);
}
__device__ __forceinline__ float fkey_dec(unsigned int k) {
    unsigned int u = (k & 0x80000000u) ? (k & 0x7FFFFFFFu) : ~k;
    return __uint_as_float(u);
}

// ---------------- K0: init scratch (acc=0, denom=0, mkey=0, h1=b1) --------
__global__ void k0_init(float* __restrict__ acc, float* __restrict__ denom,
                        unsigned int* __restrict__ mkey, float* __restrict__ h1,
                        const float* __restrict__ b1) {
    int idx = blockIdx.x * blockDim.x + threadIdx.x;
    const int A = N_NODES * OUT_D;
    if (idx < A)                       { acc[idx] = 0.f; return; }
    idx -= A;
    if (idx < N_NODES)                 { denom[idx] = 0.f; return; }
    idx -= N_NODES;
    if (idx < N_NODES)                 { mkey[idx] = 0u; return; }
    idx -= N_NODES;
    if (idx < F_DIM * EMB)             { h1[idx] = b1[idx & (EMB - 1)]; }
}

// ---------------- K1: h = x @ W_gat ; a_src/a_dst = h . att -------------
__global__ void __launch_bounds__(64)
k1_gat_proj(const float* __restrict__ x, const float* __restrict__ Wg,
            const float* __restrict__ att_s, const float* __restrict__ att_d,
            float* __restrict__ h, float* __restrict__ a_src, float* __restrict__ a_dst) {
    __shared__ float xs[F_DIM];
    __shared__ float rs[OUT_D], rd[OUT_D];
    const int node = blockIdx.x;
    const int tid  = threadIdx.x;
    for (int i = tid; i < F_DIM; i += 64) xs[i] = x[(size_t)node * F_DIM + i];
    __syncthreads();
    float acc = 0.f;
    #pragma unroll 8
    for (int k = 0; k < F_DIM; ++k) acc += xs[k] * Wg[k * OUT_D + tid];
    h[(size_t)node * OUT_D + tid] = acc;
    rs[tid] = acc * att_s[tid];
    rd[tid] = acc * att_d[tid];
    __syncthreads();
    for (int s = 32; s > 0; s >>= 1) {
        if (tid < s) { rs[tid] += rs[tid + s]; rd[tid] += rd[tid + s]; }
        __syncthreads();
    }
    if (tid == 0) { a_src[node] = rs[0]; a_dst[node] = rd[0]; }
}

// ---------------- K2: segment max of leaky_relu scores ------------------
__global__ void k2_edge_max(const long long* __restrict__ ei,
                            const float* __restrict__ a_src, const float* __restrict__ a_dst,
                            unsigned int* __restrict__ mkey) {
    int e = blockIdx.x * blockDim.x + threadIdx.x;
    if (e >= TOT_E) return;
    int s, d;
    if (e < N_EDGE) { s = (int)ei[e]; d = (int)ei[N_EDGE + e]; }
    else            { s = e - N_EDGE; d = s; }
    float es = a_src[s] + a_dst[d];
    es = es < 0.f ? 0.5f * es : es;          // GAT negative_slope = 0.5
    atomicMax(&mkey[d], fkey(es));
}

// ---------------- K3: acc[dst] += p*h[src], denom[dst] += p -------------
__global__ void __launch_bounds__(256)
k3_edge_agg(const long long* __restrict__ ei,
            const float* __restrict__ a_src, const float* __restrict__ a_dst,
            const unsigned int* __restrict__ mkey, const float* __restrict__ h,
            float* __restrict__ acc, float* __restrict__ denom) {
    long long t = (long long)blockIdx.x * blockDim.x + threadIdx.x;
    int e    = (int)(t >> 5);
    int lane = (int)(t & 31);
    if (e >= TOT_E) return;
    int s, d;
    if (e < N_EDGE) { s = (int)ei[e]; d = (int)ei[N_EDGE + e]; }
    else            { s = e - N_EDGE; d = s; }
    float es = a_src[s] + a_dst[d];
    es = es < 0.f ? 0.5f * es : es;
    float p = __expf(es - fkey_dec(mkey[d]));
    if (lane == 0) atomicAdd(&denom[d], p);
    atomicAdd(&acc[d * OUT_D + lane],      p * h[(size_t)s * OUT_D + lane]);
    atomicAdd(&acc[d * OUT_D + lane + 32], p * h[(size_t)s * OUT_D + lane + 32]);
}

// ---------------- K4: embed = lrelu(acc/denom + b); split bf16 hi/lo ----
// Writes A-side row-major [N,64] (hi/lo) and B-side K-pair-packed E^T.
__global__ void k4_embed_split(const float* __restrict__ acc, const float* __restrict__ denom,
                               const float* __restrict__ b_gat,
                               uint16_t* __restrict__ Ahi, uint16_t* __restrict__ Alo,
                               uint16_t* __restrict__ Bhi, uint16_t* __restrict__ Blo) {
    int idx = blockIdx.x * blockDim.x + threadIdx.x;   // N*64 threads
    int node = idx >> 6, c = idx & 63;
    float v = acc[idx] / denom[node] + b_gat[c];
    v = v < 0.f ? 0.01f * v : v;                       // F.leaky_relu default
    uint16_t hi = f2bf(v);
    uint16_t lo = f2bf(v - bf2f(hi));
    Ahi[idx] = hi;  Alo[idx] = lo;
    // packed: u32 word (c/2, node) = {k=2kk lo16, k=2kk+1 hi16}
    size_t bo = (size_t)(c >> 1) * (2 * N_NODES) + (size_t)node * 2 + (c & 1);
    Bhi[bo] = hi;   Blo[bo] = lo;
}

// ---------------- K5: h1 += x^T @ W1 (K-split, atomics) -----------------
__global__ void __launch_bounds__(256)
k5_h1(const float* __restrict__ x, const float* __restrict__ W1, float* __restrict__ h1) {
    __shared__ float xs[256];                 // 16 n  x 16 f
    const int f0 = blockIdx.y * 16;
    const int n0 = blockIdx.x * 2048;
    const int tid = threadIdx.x;
    const int e = tid & 127, g = tid >> 7;    // g selects 8-feature group
    float reg[8] = {0.f, 0.f, 0.f, 0.f, 0.f, 0.f, 0.f, 0.f};
    for (int nb = 0; nb < 2048; nb += 16) {
        __syncthreads();
        xs[tid] = x[(size_t)(n0 + nb + (tid >> 4)) * F_DIM + f0 + (tid & 15)];
        __syncthreads();
        #pragma unroll 4
        for (int nn = 0; nn < 16; ++nn) {
            float w = W1[(size_t)(n0 + nb + nn) * EMB + e];
            #pragma unroll
            for (int r = 0; r < 8; ++r) reg[r] += xs[nn * 16 + g * 8 + r] * w;
        }
    }
    #pragma unroll
    for (int r = 0; r < 8; ++r)
        atomicAdd(&h1[(f0 + g * 8 + r) * EMB + e], reg[r]);
}

// ---------------- K6: h2 = relu(h1) @ W2 + b2 ; store h2^T split bf16 ---
__global__ void k6_h2(const float* __restrict__ h1, const float* __restrict__ W2,
                      const float* __restrict__ b2,
                      uint16_t* __restrict__ H2hi, uint16_t* __restrict__ H2lo) {
    int idx = blockIdx.x * blockDim.x + threadIdx.x;   // 512*64
    int n = idx >> 6, k = idx & 63;
    float s = b2[k];
    #pragma unroll 8
    for (int e = 0; e < EMB; ++e) {
        float v = h1[n * EMB + e];
        v = v > 0.f ? v : 0.f;
        s += v * W2[e * OUT_D + k];
    }
    uint16_t hi = f2bf(s);
    uint16_t lo = f2bf(s - bf2f(hi));
    size_t o = (size_t)(k >> 1) * (2 * F_DIM) + n * 2 + (k & 1);
    H2hi[o] = hi;  H2lo[o] = lo;
}

// A-fragment loader: row-major bf16 [*,64] viewed as u32 (k-pairs)
__device__ __forceinline__ void load_a(FragBF& a, const uint32_t* __restrict__ base,
                                       int m, int ks, int half) {
    #pragma unroll
    for (int v = 0; v < 8; ++v)
        a.u[v] = base[m * 32 + ks * 16 + v + 4 * half + (v >= 4 ? 4 : 0)];
}
// B-fragment loader: K-pair-packed u32 [32, Ncols]
__device__ __forceinline__ void load_b(FragBF& b, const uint32_t* __restrict__ base,
                                       int n, int ks, int half, int ncols) {
    #pragma unroll
    for (int v = 0; v < 8; ++v)
        b.u[v] = base[(size_t)(ks * 16 + v + 4 * half + (v >= 4 ? 4 : 0)) * ncols + n];
}

// ---------------- K7: X_hat = embed @ h2^T  (split-bf16 WMMA) -----------
__global__ void __launch_bounds__(32)
k7_xhat(const uint32_t* __restrict__ Ahi, const uint32_t* __restrict__ Alo,
        const uint32_t* __restrict__ H2hi, const uint32_t* __restrict__ H2lo,
        float* __restrict__ X) {
    const int lane = threadIdx.x & 31;
    const int half = lane >> 4, n = lane & 15;
    const int n0 = blockIdx.x * 16;
    const int m0 = blockIdx.y * 16;
    v8f c = {};
    #pragma unroll
    for (int ks = 0; ks < 2; ++ks) {
        FragBF ah, al, bh, bl;
        load_a(ah, Ahi, m0 + n, ks, half);
        load_a(al, Alo, m0 + n, ks, half);
        load_b(bh, H2hi, n0 + n, ks, half, F_DIM);
        load_b(bl, H2lo, n0 + n, ks, half, F_DIM);
        c = __builtin_amdgcn_wmma_f32_16x16x32_bf16(false, ah.v, false, bh.v, (short)0, c, false, false);
        c = __builtin_amdgcn_wmma_f32_16x16x32_bf16(false, ah.v, false, bl.v, (short)0, c, false, false);
        c = __builtin_amdgcn_wmma_f32_16x16x32_bf16(false, al.v, false, bh.v, (short)0, c, false, false);
    }
    #pragma unroll
    for (int r = 0; r < 8; ++r)
        X[(size_t)(m0 + r + 8 * half) * F_DIM + n0 + n] = c[r];
}

// ---------------- K8: A_hat = sigmoid(E @ E^T)  (dominant kernel) -------
__global__ void __launch_bounds__(256)
k8_ahat(const uint32_t* __restrict__ Ahi, const uint32_t* __restrict__ Alo,
        const uint32_t* __restrict__ Bhi, const uint32_t* __restrict__ Blo,
        float* __restrict__ Aout) {
    const int lane = threadIdx.x & 31;
    const int w    = threadIdx.x >> 5;          // 8 waves: 2 (rows) x 4 (cols)
    const int half = lane >> 4, n = lane & 15;
    const int m0 = blockIdx.y * 64  + (w >> 2) * 32;
    const int n0 = blockIdx.x * 128 + (w & 3) * 32;
    v8f c[2][2] = {};
    #pragma unroll
    for (int ks = 0; ks < 2; ++ks) {
        FragBF ah[2], al[2], bh[2], bl[2];
        #pragma unroll
        for (int i = 0; i < 2; ++i) {
            load_a(ah[i], Ahi, m0 + i * 16 + n, ks, half);
            load_a(al[i], Alo, m0 + i * 16 + n, ks, half);
            load_b(bh[i], Bhi, n0 + i * 16 + n, ks, half, N_NODES);
            load_b(bl[i], Blo, n0 + i * 16 + n, ks, half, N_NODES);
        }
        #pragma unroll
        for (int i = 0; i < 2; ++i)
            #pragma unroll
            for (int j = 0; j < 2; ++j) {
                c[i][j] = __builtin_amdgcn_wmma_f32_16x16x32_bf16(false, ah[i].v, false, bh[j].v, (short)0, c[i][j], false, false);
                c[i][j] = __builtin_amdgcn_wmma_f32_16x16x32_bf16(false, ah[i].v, false, bl[j].v, (short)0, c[i][j], false, false);
                c[i][j] = __builtin_amdgcn_wmma_f32_16x16x32_bf16(false, al[i].v, false, bh[j].v, (short)0, c[i][j], false, false);
            }
    }
    #pragma unroll
    for (int i = 0; i < 2; ++i)
        #pragma unroll
        for (int j = 0; j < 2; ++j)
            #pragma unroll
            for (int r = 0; r < 8; ++r) {
                float v = c[i][j][r];
                float sg = 1.0f / (1.0f + __expf(-v));
                size_t off = (size_t)(m0 + 16 * i + r + 8 * half) * N_NODES + n0 + 16 * j + n;
                __builtin_nontemporal_store(sg, Aout + off);   // 1 GiB stream: bypass temporal caching
            }
}

// ------------------------------------------------------------------------
extern "C" void kernel_launch(void* const* d_in, const int* in_sizes, int n_in,
                              void* d_out, int out_size, void* d_ws, size_t ws_size,
                              hipStream_t stream) {
    const float*     x     = (const float*)d_in[0];
    const long long* ei    = (const long long*)d_in[1];
    // d_in[2] = adj (unused)
    const float*     Wg    = (const float*)d_in[3];
    const float*     att_s = (const float*)d_in[4];
    const float*     att_d = (const float*)d_in[5];
    const float*     b_gat = (const float*)d_in[6];
    const float*     W1    = (const float*)d_in[7];
    const float*     b1    = (const float*)d_in[8];
    const float*     W2    = (const float*)d_in[9];
    const float*     b2    = (const float*)d_in[10];

    char* ws = (char*)d_ws;
    float*        h     = (float*)(ws + 0);                       // 4 MiB
    float*        acc   = (float*)(ws + (4u << 20));              // 4 MiB
    float*        a_src = (float*)(ws + (8u << 20));              // 64 KiB
    float*        a_dst = (float*)(ws + (8u << 20) + (64u << 10));
    float*        denom = (float*)(ws + (8u << 20) + (128u << 10));
    unsigned int* mkey  = (unsigned int*)(ws + (8u << 20) + (192u << 10));
    float*        h1    = (float*)(ws + (8u << 20) + (256u << 10)); // 256 KiB
    uint16_t*     Ahi   = (uint16_t*)(ws + (9u << 20));           // 2 MiB each
    uint16_t*     Alo   = (uint16_t*)(ws + (11u << 20));
    uint16_t*     Bhi   = (uint16_t*)(ws + (13u << 20));
    uint16_t*     Blo   = (uint16_t*)(ws + (15u << 20));
    uint16_t*     H2hi  = (uint16_t*)(ws + (17u << 20));          // 64 KiB each
    uint16_t*     H2lo  = (uint16_t*)(ws + (17u << 20) + (64u << 10));

    float* A_hat = (float*)d_out;
    float* X_hat = (float*)d_out + (size_t)N_NODES * N_NODES;

    // K0: init accumulators / h1 bias
    {
        int total = N_NODES * OUT_D + 2 * N_NODES + F_DIM * EMB;
        k0_init<<<(total + 255) / 256, 256, 0, stream>>>(acc, denom, mkey, h1, b1);
    }
    // K1: GAT linear projection + attention logits
    k1_gat_proj<<<N_NODES, 64, 0, stream>>>(x, Wg, att_s, att_d, h, a_src, a_dst);
    // K2: per-dst segment max
    k2_edge_max<<<TOT_E / 256, 256, 0, stream>>>(ei, a_src, a_dst, mkey);
    // K3: softmax numerator aggregation (wave per edge, 2 channels/lane)
    k3_edge_agg<<<(TOT_E * 32) / 256, 256, 0, stream>>>(ei, a_src, a_dst, mkey, h, acc, denom);
    // K4: embed = lrelu(acc/denom + b); split into bf16 hi/lo, row + packed-T layouts
    k4_embed_split<<<(N_NODES * OUT_D) / 256, 256, 0, stream>>>(acc, denom, b_gat, Ahi, Alo, Bhi, Blo);
    // K5: h1 = x^T @ W1 (+b1 from init), K split 8 ways
    k5_h1<<<dim3(8, 32), 256, 0, stream>>>(x, W1, h1);
    // K6: h2 = relu(h1) @ W2 + b2, stored transposed, split bf16
    k6_h2<<<(F_DIM * OUT_D) / 256, 256, 0, stream>>>(h1, W2, b2, H2hi, H2lo);
    // K7: X_hat = embed @ h2^T (WMMA)
    k7_xhat<<<dim3(F_DIM / 16, N_NODES / 16), 32, 0, stream>>>(
        (const uint32_t*)Ahi, (const uint32_t*)Alo,
        (const uint32_t*)H2hi, (const uint32_t*)H2lo, X_hat);
    // K8: A_hat = sigmoid(embed @ embed^T) (WMMA, NT stores) — dominant
    k8_ahat<<<dim3(N_NODES / 128, N_NODES / 64), 256, 0, stream>>>(
        (const uint32_t*)Ahi, (const uint32_t*)Alo,
        (const uint32_t*)Bhi, (const uint32_t*)Blo, A_hat);
}
